// OuterProductMeanSS_28733331210481
// MI455X (gfx1250) — matrix operand order repaired and checked by reading, not compiled, not run on hardware
//
#include <hip/hip_runtime.h>
#include <hip/hip_bf16.h>

// OuterProductMean: z[i,j,:] = sum_{c,e} a[i,c] b[j,e] Wr[c,e,:] + b_out
// Factorized: T[j,c,z] = sum_e b[j,e] Wr[c,e,z]   (512x4096 GEMM, K=32)
//             z[i,j,z] = sum_c a[i,c] T[j,c,z]    (per-j 512x128 GEMM, K=32)
// Compute via v_wmma_f32_16x16x32_bf16; output write (134 MB) is the roofline.

typedef __attribute__((ext_vector_type(16))) __bf16 v16bf;
typedef __attribute__((ext_vector_type(8)))  __bf16 v8bf;
typedef __attribute__((ext_vector_type(8)))  float  v8f;

#define LSIZE 512
#define CS 384
#define CH 32
#define CZ 128

// ---------------- Stage 1: LayerNorm + projections + mask -> bf16 a,b -------
__global__ void opm_stage1(const float* __restrict__ s, const int* __restrict__ mask,
                           const float* __restrict__ ln_scale, const float* __restrict__ ln_bias,
                           const float* __restrict__ w1, const float* __restrict__ b1,
                           const float* __restrict__ w2, const float* __restrict__ b2,
                           __bf16* __restrict__ a_bf, __bf16* __restrict__ b_bf) {
  __shared__ float sn[CS];
  __shared__ float red[128];
  const int i = blockIdx.x;          // row 0..511
  const int t = threadIdx.x;         // 0..127
  const float* srow = s + i * CS;
  float v0 = srow[t], v1 = srow[t + 128], v2 = srow[t + 256];
  red[t] = v0 + v1 + v2;
  __syncthreads();
  for (int off = 64; off > 0; off >>= 1) {
    if (t < off) red[t] += red[t + off];
    __syncthreads();
  }
  float mu = red[0] * (1.0f / (float)CS);
  __syncthreads();
  float d0 = v0 - mu, d1 = v1 - mu, d2 = v2 - mu;
  red[t] = d0 * d0 + d1 * d1 + d2 * d2;
  __syncthreads();
  for (int off = 64; off > 0; off >>= 1) {
    if (t < off) red[t] += red[t + off];
    __syncthreads();
  }
  float inv = rsqrtf(red[0] * (1.0f / (float)CS) + 1e-5f);
  sn[t]       = d0 * inv * ln_scale[t]       + ln_bias[t];
  sn[t + 128] = d1 * inv * ln_scale[t + 128] + ln_bias[t + 128];
  sn[t + 256] = d2 * inv * ln_scale[t + 256] + ln_bias[t + 256];
  __syncthreads();
  if (t < 64) {
    const int h = t & 31;
    const float* W = (t < 32) ? w1 : w2;
    float acc = (t < 32) ? b1[h] : b2[h];
    for (int k = 0; k < CS; ++k) acc += sn[k] * W[k * CH + h];
    acc *= (float)mask[i];
    __bf16 r = (__bf16)acc;
    if (t < 32) a_bf[i * CH + h] = r;
    else        b_bf[i * CH + h] = r;
  }
}

// ------------- Repack w_out[(c*32+e)*128+z] -> W2t[(z*32+c)*32+e] bf16 ------
__global__ void opm_prep_w2t(const float* __restrict__ w_out, __bf16* __restrict__ w2t) {
  int tid = blockIdx.x * 256 + threadIdx.x;        // 131072 total
  int e = tid & 31, c = (tid >> 5) & 31, z = tid >> 10;
  w2t[tid] = (__bf16)w_out[(c * 32 + e) * CZ + z]; // out idx == (z*32+c)*32+e == tid
}

// ---- Stage 2: Tt[j][z*32+c] = sum_e b[j,e] * W2t[(z*32+c)*32+e]  (WMMA) ----
__global__ void opm_stage2(const __bf16* __restrict__ bvec, const __bf16* __restrict__ w2t,
                           __bf16* __restrict__ tt) {
  const int lane = threadIdx.x & 31;
  const int wave = threadIdx.x >> 5;
  const int col_tile = blockIdx.x * 8 + wave;  // 0..255 (cols of 4096/16)
  const int j_tile   = blockIdx.y;             // 0..31
  const int lm = lane & 15;
  const int hi = lane >> 4;                    // half-wave select
  // A fragment (16x32 bf16): lane holds row m, K = k0..k0+7 and k0+16..k0+23
  const int row = j_tile * 16 + lm;
  const int ka = hi ? 8 : 0;
  union { v16bf v; v8bf h[2]; } af;
  af.h[0] = *(const v8bf*)(bvec + row * CH + ka);
  af.h[1] = *(const v8bf*)(bvec + row * CH + ka + 16);
  // B fragment (32x16 bf16): lane holds col, K = kb..kb+7 and kb+16..kb+23
  const int col = col_tile * 16 + lm;
  const int kb = hi ? 8 : 0;
  union { v16bf v; v8bf h[2]; } bfg;
  bfg.h[0] = *(const v8bf*)(w2t + col * CH + kb);
  bfg.h[1] = *(const v8bf*)(w2t + col * CH + kb + 16);
  v8f acc = {};
  acc = __builtin_amdgcn_wmma_f32_16x16x32_bf16(false, af.v, false, bfg.v,
                                                (short)0, acc, false, false);
#pragma unroll
  for (int v = 0; v < 8; ++v) {
    int r = j_tile * 16 + v + hi * 8;          // C layout: VGPR v -> rows v / v+8
    tt[r * 4096 + col] = (__bf16)acc[v];
  }
}

// ---- Stage 3: z[i,j,zc] = sum_c a[i,c] * Tt[j][zc_tile..][c] + b_out -------
__global__ void opm_stage3(const __bf16* __restrict__ avec, const __bf16* __restrict__ tt,
                           const float* __restrict__ bout, float* __restrict__ out) {
  const int lane = threadIdx.x & 31;
  const int wave = threadIdx.x >> 5;
  const int i_tile = blockIdx.x * 8 + wave;    // 0..31
  const int j = blockIdx.y;                    // 0..511
  const int lm = lane & 15;
  const int hi = lane >> 4;
  // A fragment from a (16 rows x 32 K)
  const int row = i_tile * 16 + lm;
  const int ka = hi ? 8 : 0;
  union { v16bf v; v8bf h[2]; } af;
  af.h[0] = *(const v8bf*)(avec + row * CH + ka);
  af.h[1] = *(const v8bf*)(avec + row * CH + ka + 16);
  const __bf16* tj = tt + j * 4096;            // Tt[j][z*32+c]
  const int kb = hi ? 8 : 0;
#pragma unroll
  for (int zt = 0; zt < 8; ++zt) {
    const int zc = zt * 16 + lm;               // output channel this lane owns
    union { v16bf v; v8bf h[2]; } bfg;         // B: K=c at Tt[j][zc*32+kb..+7] and [+16..+23]
    bfg.h[0] = *(const v8bf*)(tj + zc * CH + kb);
    bfg.h[1] = *(const v8bf*)(tj + zc * CH + kb + 16);
    const float bz = bout[zc];                 // bias depends only on column
    v8f acc = { bz, bz, bz, bz, bz, bz, bz, bz };
    acc = __builtin_amdgcn_wmma_f32_16x16x32_bf16(false, af.v, false, bfg.v,
                                                  (short)0, acc, false, false);
#pragma unroll
    for (int v = 0; v < 8; ++v) {
      long i = (long)(i_tile * 16 + v + hi * 8);
      out[((i * LSIZE + j) << 7) + zc] = acc[v];
    }
  }
}

extern "C" void kernel_launch(void* const* d_in, const int* in_sizes, int n_in,
                              void* d_out, int out_size, void* d_ws, size_t ws_size,
                              hipStream_t stream) {
  const float* s        = (const float*)d_in[0];
  const int*   mask     = (const int*)  d_in[1];
  const float* ln_scale = (const float*)d_in[2];
  const float* ln_bias  = (const float*)d_in[3];
  const float* w1       = (const float*)d_in[4];
  const float* b1       = (const float*)d_in[5];
  const float* w2       = (const float*)d_in[6];
  const float* b2       = (const float*)d_in[7];
  const float* w_out    = (const float*)d_in[8];
  const float* b_out    = (const float*)d_in[9];
  float* out = (float*)d_out;

  char* ws = (char*)d_ws;
  __bf16* a_bf = (__bf16*)(ws);                         // 512*32*2   = 32 KB
  __bf16* b_bf = (__bf16*)(ws + (32 << 10));            // 32 KB
  __bf16* w2t  = (__bf16*)(ws + (64 << 10));            // 32*4096*2  = 256 KB
  __bf16* tt   = (__bf16*)(ws + (64 << 10) + (256 << 10)); // 512*4096*2 = 4 MB

  opm_stage1<<<LSIZE, 128, 0, stream>>>(s, mask, ln_scale, ln_bias, w1, b1, w2, b2, a_bf, b_bf);
  opm_prep_w2t<<<512, 256, 0, stream>>>(w_out, w2t);
  opm_stage2<<<dim3(32, 32), 256, 0, stream>>>(b_bf, w2t, tt);
  opm_stage3<<<dim3(4, LSIZE), 256, 0, stream>>>(a_bf, tt, b_out, out);
}
